// CausalSelfAttention_7997229105934
// MI455X (gfx1250) — matrix operand order repaired and checked by reading, not compiled, not run on hardware
//
#include <hip/hip_runtime.h>
#include <hip/hip_bf16.h>
#include <stdint.h>
#include <stddef.h>

// ---------------------------------------------------------------------------
// CausalSelfAttention (GQA) for gfx1250 / MI455X, bf16 WMMA everywhere.
// B=2, T=2048, C=2048, NH=16, NKV=4, HD=128, QKV_N=3072.
// ---------------------------------------------------------------------------

typedef __bf16 bf16_t;
typedef __attribute__((ext_vector_type(16))) __bf16 v16bf;
typedef __attribute__((ext_vector_type(8)))  float  v8f;
typedef int vi4_t __attribute__((ext_vector_type(4)));

#define WMMA_BF16(A, B, C) \
  __builtin_amdgcn_wmma_f32_16x16x32_bf16(false, (A), false, (B), (short)0, (C), false, false)

static constexpr int B_  = 2;
static constexpr int T_  = 2048;
static constexpr int C_  = 2048;
static constexpr int NH  = 16;
static constexpr int NKV = 4;
static constexpr int HD  = 128;
static constexpr int QKV_N = (NH + 2 * NKV) * HD; // 3072
static constexpr int M_  = B_ * T_;               // 4096

// --- CDNA5 async global->LDS copy (ASYNCcnt path), with safe fallback -------
#if defined(__has_builtin)
#if __has_builtin(__builtin_amdgcn_global_load_async_to_lds_b128) && \
    __has_builtin(__builtin_amdgcn_s_wait_asynccnt)
#define USE_ASYNC_LDS 1
#endif
#endif
#ifndef USE_ASYNC_LDS
#define USE_ASYNC_LDS 0
#endif

__device__ __forceinline__ void copy16_g2l(const bf16_t* g, bf16_t* l) {
#if USE_ASYNC_LDS
  __builtin_amdgcn_global_load_async_to_lds_b128(
      (__attribute__((address_space(1))) vi4_t*)g,
      (__attribute__((address_space(3))) vi4_t*)l, 0, 0);
#else
  *reinterpret_cast<int4*>(l) = *reinterpret_cast<const int4*>(g);
#endif
}
__device__ __forceinline__ void async_fence() {
#if USE_ASYNC_LDS
  __builtin_amdgcn_s_wait_asynccnt(0);
#endif
}

// Build a v16bf fragment from two 16-byte (8 x bf16) chunks.
__device__ __forceinline__ v16bf frag2(const bf16_t* p0, const bf16_t* p1) {
  union { v16bf v; int4 q[2]; } u;
  u.q[0] = *reinterpret_cast<const int4*>(p0);
  u.q[1] = *reinterpret_cast<const int4*>(p1);
  return u.v;
}

// ---------------------------------------------------------------------------
// f32 -> bf16 conversion, 4 elems/thread (bandwidth-bound)
// ---------------------------------------------------------------------------
__global__ void cvt_f32_bf16_kernel(const float* __restrict__ in,
                                    bf16_t* __restrict__ out, int n4) {
  int i = blockIdx.x * blockDim.x + threadIdx.x;
  if (i >= n4) return;
  const float4 v = reinterpret_cast<const float4*>(in)[i];
  union { bf16_t h[4]; uint64_t u; } p;
  p.h[0] = (bf16_t)v.x; p.h[1] = (bf16_t)v.y;
  p.h[2] = (bf16_t)v.z; p.h[3] = (bf16_t)v.w;
  reinterpret_cast<uint64_t*>(out)[i] = p.u;
}

// ---------------------------------------------------------------------------
// GEMM: A[M,K] bf16 row-major  x  Bm[K,N] bf16 row-major  + bias[N]
// Block tile 128(M) x 128(N), 8 waves (256 thr), K-step 32.
// Each wave: 16 rows x 128 cols = 8 WMMA accumulators (8 WMMAs per A-frag).
// ---------------------------------------------------------------------------
template <bool OUT_BF16>
__global__ __launch_bounds__(256)
void gemm_bf16_kernel(const bf16_t* __restrict__ A, const bf16_t* __restrict__ Bm,
                      const float* __restrict__ bias, void* __restrict__ Out,
                      int M, int N, int K) {
  __shared__ bf16_t aT[128][40]; // [m][k], +8 pad (row = 80B, 16B aligned)
  __shared__ bf16_t bT[128][40]; // [n][k] (transposed), +8 pad

  const int tid  = threadIdx.x;
  const int wave = tid >> 5;
  const int lane = tid & 31;
  const int lh   = lane >> 4;   // lane-half
  const int l16  = lane & 15;

  const int m0 = blockIdx.y * 128;
  const int n0 = blockIdx.x * 128;

  v8f acc[8];
#pragma unroll
  for (int nt = 0; nt < 8; ++nt)
#pragma unroll
    for (int r = 0; r < 8; ++r) acc[nt][r] = 0.0f;

  auto loadB = [&](int nt) {
    const bf16_t* bp = &bT[nt * 16 + l16][lh * 8];
    return frag2(bp, bp + 16);
  };

  for (int k0 = 0; k0 < K; k0 += 32) {
    __syncthreads();
    // A tile 128x32 (512 x 16B chunks; 2 per thread): async global->LDS.
#pragma unroll
    for (int i = 0; i < 2; ++i) {
      int idx = i * 256 + tid;
      int row = idx >> 2;       // 4 chunks of 8 elems per 32-wide row
      int c8  = idx & 3;
      copy16_g2l(A + (size_t)(m0 + row) * K + k0 + c8 * 8, &aT[row][c8 * 8]);
    }
    // B tile 32x128 -> transposed [n][k] in LDS (2048 x u32; 8 per thread).
#pragma unroll
    for (int i = 0; i < 8; ++i) {
      int u  = i * 256 + tid;
      int k  = u >> 6;          // 64 u32 per k-row
      int np = u & 63;          // n-pair
      uint32_t w =
          *reinterpret_cast<const uint32_t*>(Bm + (size_t)(k0 + k) * N + n0 + np * 2);
      union { uint32_t w; bf16_t h[2]; } cv; cv.w = w;
      bT[np * 2 + 0][k] = cv.h[0];
      bT[np * 2 + 1][k] = cv.h[1];
    }
    async_fence();
    __syncthreads();

    // A fragment: lane row m = wave*16 + l16; K = {8*lh..} and {16+8*lh..}
    const bf16_t* ap = &aT[wave * 16 + l16][lh * 8];
    v16bf af = frag2(ap, ap + 16);
    // Software-pipelined n-tile loop: fetch B(nt+1) before issuing WMMA(nt).
    v16bf bcur = loadB(0);
#pragma unroll
    for (int nt = 0; nt < 8; ++nt) {
      v16bf bnext = (nt < 7) ? loadB(nt + 1) : bcur;
      acc[nt] = WMMA_BF16(af, bcur, acc[nt]);
      bcur = bnext;
    }
  }

#pragma unroll
  for (int nt = 0; nt < 8; ++nt)
#pragma unroll
    for (int r = 0; r < 8; ++r) {
      int m = m0 + wave * 16 + r + 8 * lh; // C/D layout: VGPR r -> row r + 8*lh
      int n = n0 + nt * 16 + l16;
      float v = acc[nt][r] + bias[n];
      if (OUT_BF16)
        reinterpret_cast<bf16_t*>(Out)[(size_t)m * N + n] = (bf16_t)v;
      else
        reinterpret_cast<float*>(Out)[(size_t)m * N + n] = v;
    }
}

// ---------------------------------------------------------------------------
// RoPE + reformat: qkv[B*T][3072] bf16 -> qh[B][16][T][128], kh[B][4][T][128]
// (rope applied), vT[B][4][128][T] (transposed, no rope).
// ---------------------------------------------------------------------------
__global__ void rope_kernel(const bf16_t* __restrict__ qkv,
                            const float* __restrict__ rope,
                            bf16_t* __restrict__ qh, bf16_t* __restrict__ khd,
                            bf16_t* __restrict__ vtv) {
  int idx = blockIdx.x * blockDim.x + threadIdx.x;
  int i   = idx & 63;           // pair index within head (d = 2i, 2i+1)
  int hh  = (idx >> 6) % 24;    // 0..15 q heads, 16..19 k heads, 20..23 v heads
  int row = idx / (64 * 24);    // b*T + t
  if (row >= M_) return;
  int b = row >> 11, t = row & 2047;

  const bf16_t* src = qkv + (size_t)row * QKV_N + hh * HD + 2 * i;
  float x0 = (float)src[0], x1 = (float)src[1];

  if (hh < 20) {
    float s = rope[t * HD + 2 * i];      // rope[:,0::2] = sin
    float c = rope[t * HD + 2 * i + 1];  // rope[:,1::2] = cos
    float o0 = x0 * c - x1 * s;
    float o1 = x1 * c + x0 * s;
    if (hh < 16) {
      bf16_t* dst = qh + ((size_t)(b * NH + hh) * T_ + t) * HD + 2 * i;
      dst[0] = (bf16_t)o0; dst[1] = (bf16_t)o1;
    } else {
      int kvh = hh - 16;
      bf16_t* dst = khd + ((size_t)(b * NKV + kvh) * T_ + t) * HD + 2 * i;
      dst[0] = (bf16_t)o0; dst[1] = (bf16_t)o1;
    }
  } else {
    int kvh = hh - 20;
    // vT[b][kvh][d][t]
    vtv[((size_t)(b * NKV + kvh) * HD + 2 * i) * T_ + t]     = (bf16_t)x0;
    vtv[((size_t)(b * NKV + kvh) * HD + 2 * i + 1) * T_ + t] = (bf16_t)x1;
  }
}

// ---------------------------------------------------------------------------
// Flash attention: block = (qtile of 64 rows, head, batch); 4 waves x 16 rows.
// y written as [B][T][NH*HD] bf16 (row-major input for the out-projection).
// ---------------------------------------------------------------------------
__global__ __launch_bounds__(128)
void attn_kernel(const bf16_t* __restrict__ qh, const bf16_t* __restrict__ khd,
                 const bf16_t* __restrict__ vtv, bf16_t* __restrict__ y) {
  __shared__ bf16_t k_lds[64][HD + 8];   // [key][d]
  __shared__ bf16_t v_lds[HD][64 + 8];   // [d][key]   (from vT)
  __shared__ bf16_t p_lds[4][16][64 + 8];// per-wave P [m][k]

  const int qt = blockIdx.x, h = blockIdx.y, b = blockIdx.z;
  const int kvh = h / (NH / NKV);

  const int tid  = threadIdx.x;
  const int wave = tid >> 5;
  const int lane = tid & 31;
  const int lh   = lane >> 4;
  const int l16  = lane & 15;

  // Q fragments for this wave's 16 rows (reused over all key tiles).
  const bf16_t* qbase =
      qh + ((size_t)(b * NH + h) * T_ + qt * 64 + wave * 16) * HD;
  v16bf qf[4];
#pragma unroll
  for (int dc = 0; dc < 4; ++dc) {
    const bf16_t* p0 = qbase + (size_t)l16 * HD + dc * 32 + lh * 8;
    qf[dc] = frag2(p0, p0 + 16);
  }

  v8f o[8];
#pragma unroll
  for (int dt = 0; dt < 8; ++dt)
#pragma unroll
    for (int r = 0; r < 8; ++r) o[dt][r] = 0.0f;
  float mrow[8], lrow[8];
#pragma unroll
  for (int r = 0; r < 8; ++r) { mrow[r] = -1e30f; lrow[r] = 0.0f; }

  const bf16_t* kbase_g = khd + (size_t)(b * NKV + kvh) * T_ * HD;
  const bf16_t* vbase_g = vtv + (size_t)(b * NKV + kvh) * HD * T_;

  const float sc = 1.0f / 128.0f; // (1/sqrt(HD))^2 folded into logits
  const int nkt = qt + 1;         // causal: key tiles covering keys <= qmax

  auto loadKB = [&](int nt, int dc) {
    const bf16_t* bp = &k_lds[nt * 16 + l16][dc * 32 + lh * 8];
    return frag2(bp, bp + 16);
  };
  auto loadVB = [&](int kc, int dt) {
    const bf16_t* bp = &v_lds[dt * 16 + l16][kc * 32 + lh * 8];
    return frag2(bp, bp + 16);
  };

  for (int kt = 0; kt < nkt; ++kt) {
    const int k0 = kt * 64;
    __syncthreads(); // protect LDS tiles from previous iteration's readers

    // K tile 64x128 (1024 x 16B chunks; 8 per thread): async global->LDS.
#pragma unroll
    for (int i = 0; i < 8; ++i) {
      int idx = i * 128 + tid;
      int row = idx >> 4;  // 16 chunks per 128-wide row
      int c   = idx & 15;
      copy16_g2l(kbase_g + (size_t)(k0 + row) * HD + c * 8, &k_lds[row][c * 8]);
    }
    // V^T tile 128x64 (1024 x 16B chunks; 8 per thread): async global->LDS.
#pragma unroll
    for (int i = 0; i < 8; ++i) {
      int idx = i * 128 + tid;
      int row = idx >> 3;  // 8 chunks per 64-wide row
      int c   = idx & 7;
      copy16_g2l(vbase_g + (size_t)row * T_ + k0 + c * 8, &v_lds[row][c * 8]);
    }
    // Prefetch next key tile into cache while this one is consumed.
    if (kt + 1 < nkt) {
      __builtin_prefetch(kbase_g + (size_t)(k0 + 64 + (tid >> 1)) * HD +
                             (tid & 1) * 64, 0, 1);
      __builtin_prefetch(vbase_g + (size_t)tid * T_ + k0 + 64, 0, 1);
    }
    async_fence();
    __syncthreads();

    // S = Q @ K^T : 4 n-tiles x 4 d-chunks of WMMA (pipelined B-frag loads)
    v8f s[4];
    {
      v16bf bcur = loadKB(0, 0);
#pragma unroll
      for (int nt = 0; nt < 4; ++nt) {
        v8f a;
#pragma unroll
        for (int r = 0; r < 8; ++r) a[r] = 0.0f;
#pragma unroll
        for (int dc = 0; dc < 4; ++dc) {
          v16bf bnext = (dc < 3)   ? loadKB(nt, dc + 1)
                        : (nt < 3) ? loadKB(nt + 1, 0)
                                   : bcur;
          a = WMMA_BF16(qf[dc], bcur, a);
          bcur = bnext;
        }
        s[nt] = a;
      }
    }

    // Scale + causal mask + tile row-max
    float tmax[8];
#pragma unroll
    for (int r = 0; r < 8; ++r) tmax[r] = -1e30f;
#pragma unroll
    for (int nt = 0; nt < 4; ++nt)
#pragma unroll
      for (int r = 0; r < 8; ++r) {
        int key = k0 + nt * 16 + l16;
        int qi  = qt * 64 + wave * 16 + r + 8 * lh;
        float v = (key <= qi) ? s[nt][r] * sc : -1e30f;
        s[nt][r] = v;
        tmax[r] = fmaxf(tmax[r], v);
      }
#pragma unroll
    for (int r = 0; r < 8; ++r)
#pragma unroll
      for (int off = 8; off > 0; off >>= 1)
        tmax[r] = fmaxf(tmax[r], __shfl_xor(tmax[r], off, 32));

    // Online softmax update; rescale existing accumulators.
    float psum[8];
#pragma unroll
    for (int r = 0; r < 8; ++r) {
      float nm = fmaxf(mrow[r], tmax[r]);
      float f  = __expf(mrow[r] - nm);
      mrow[r] = nm;
      lrow[r] *= f;
#pragma unroll
      for (int dt = 0; dt < 8; ++dt) o[dt][r] *= f;
      psum[r] = 0.0f;
    }
    // P = exp(S - m), staged in wave-private LDS in A-fragment order.
#pragma unroll
    for (int nt = 0; nt < 4; ++nt)
#pragma unroll
      for (int r = 0; r < 8; ++r) {
        float p = __expf(s[nt][r] - mrow[r]);
        psum[r] += p;
        p_lds[wave][r + 8 * lh][nt * 16 + l16] = (bf16_t)p;
      }
#pragma unroll
    for (int r = 0; r < 8; ++r) {
#pragma unroll
      for (int off = 8; off > 0; off >>= 1)
        psum[r] += __shfl_xor(psum[r], off, 32);
      lrow[r] += psum[r];
    }

    // O += P @ V : 2 k-chunks x 8 d-tiles of WMMA (pipelined B-frag loads)
    {
      v16bf bcur = loadVB(0, 0);
#pragma unroll
      for (int kc = 0; kc < 2; ++kc) {
        const bf16_t* ap = &p_lds[wave][l16][kc * 32 + lh * 8];
        v16bf af = frag2(ap, ap + 16);
#pragma unroll
        for (int dt = 0; dt < 8; ++dt) {
          v16bf bnext = (dt < 7)   ? loadVB(kc, dt + 1)
                        : (kc < 1) ? loadVB(kc + 1, 0)
                                   : bcur;
          o[dt] = WMMA_BF16(af, bcur, o[dt]);
          bcur = bnext;
        }
      }
    }
  }

  // Epilogue: y[b][t][h*128 + d] = O / l   (bf16, row-major for out-proj GEMM)
  bf16_t* ybase =
      y + (size_t)(b * T_ + qt * 64 + wave * 16) * C_ + h * HD;
#pragma unroll
  for (int dt = 0; dt < 8; ++dt)
#pragma unroll
    for (int r = 0; r < 8; ++r) {
      int m = r + 8 * lh;
      float v = o[dt][r] / lrow[r];
      ybase[(size_t)m * C_ + dt * 16 + l16] = (bf16_t)v;
    }
}

// ---------------------------------------------------------------------------
// Host-side orchestration
// ---------------------------------------------------------------------------
extern "C" void kernel_launch(void* const* d_in, const int* in_sizes, int n_in,
                              void* d_out, int out_size, void* d_ws, size_t ws_size,
                              hipStream_t stream) {
  const float* x     = (const float*)d_in[0];
  const float* ropec = (const float*)d_in[1];
  const float* W_qkv = (const float*)d_in[2];
  const float* b_qkv = (const float*)d_in[3];
  const float* W_out = (const float*)d_in[4];
  const float* b_out = (const float*)d_in[5];
  float* out = (float*)d_out;

  char* ws = (char*)d_ws;
  size_t off = 0;
  auto carve = [&](size_t bytes) {
    char* p = ws + off;
    off += (bytes + 255) & ~(size_t)255;
    return p;
  };
  bf16_t* xb    = (bf16_t*)carve((size_t)M_ * C_ * 2);       // 16 MB
  bf16_t* wqkvb = (bf16_t*)carve((size_t)C_ * QKV_N * 2);    // 12 MB
  bf16_t* woutb = (bf16_t*)carve((size_t)C_ * C_ * 2);       //  8 MB
  bf16_t* qkvb  = (bf16_t*)carve((size_t)M_ * QKV_N * 2);    // 24 MB
  bf16_t* q_h   = (bf16_t*)carve((size_t)B_ * NH * T_ * HD * 2);   // 16 MB
  bf16_t* k_h   = (bf16_t*)carve((size_t)B_ * NKV * T_ * HD * 2);  //  4 MB
  bf16_t* v_t   = (bf16_t*)carve((size_t)B_ * NKV * HD * T_ * 2);  //  4 MB
  bf16_t* ybuf  = (bf16_t*)carve((size_t)M_ * C_ * 2);       // 16 MB
  if (off > ws_size) return; // insufficient scratch

  // 1) f32 -> bf16 conversions (vectorized x4)
  {
    int n4 = (M_ * C_) / 4;
    cvt_f32_bf16_kernel<<<(n4 + 255) / 256, 256, 0, stream>>>(x, xb, n4);
    n4 = (C_ * QKV_N) / 4;
    cvt_f32_bf16_kernel<<<(n4 + 255) / 256, 256, 0, stream>>>(W_qkv, wqkvb, n4);
    n4 = (C_ * C_) / 4;
    cvt_f32_bf16_kernel<<<(n4 + 255) / 256, 256, 0, stream>>>(W_out, woutb, n4);
  }
  // 2) QKV GEMM: [4096,2048] x [2048,3072] + b_qkv -> bf16
  {
    dim3 grid(QKV_N / 128, M_ / 128);
    gemm_bf16_kernel<true><<<grid, 256, 0, stream>>>(xb, wqkvb, b_qkv, qkvb,
                                                     M_, QKV_N, C_);
  }
  // 3) RoPE + head-major reformat (+ V transpose)
  {
    int n = M_ * 24 * 64;
    rope_kernel<<<(n + 255) / 256, 256, 0, stream>>>(qkvb, ropec, q_h, k_h, v_t);
  }
  // 4) Flash attention
  {
    dim3 grid(T_ / 64, NH, B_);
    attn_kernel<<<grid, 128, 0, stream>>>(q_h, k_h, v_t, ybuf);
  }
  // 5) Output projection: [4096,2048] x [2048,2048] + b_out -> f32 d_out
  {
    dim3 grid(C_ / 128, M_ / 128);
    gemm_bf16_kernel<false><<<grid, 256, 0, stream>>>(ybuf, woutb, b_out, out,
                                                      M_, C_, C_);
  }
}
